// ClusteringLSTM_68753836474849
// MI455X (gfx1250) — compile-verified, matrix operand order checked by reading, not run on hardware
//
#include <hip/hip_runtime.h>

// ---------------- problem constants ----------------
#define T_    256
#define B_    32
#define E_    256
#define H_    512
#define V_    2048
#define G4H   2048            // 4*H
#define NCLUS 6
#define NPRED 10
#define TB    (T_*B_)         // 8192

// ---------------- WMMA types (CDNA5 wave32) ----------------
typedef __attribute__((ext_vector_type(16))) __bf16 v16bf;
typedef __attribute__((ext_vector_type(8)))  float  v8f;

union FragU { uint4 q[2]; v16bf v; };

__device__ __forceinline__ unsigned short f2bf(float f) {
    unsigned int u = __float_as_uint(f);
    u += 0x7FFFu + ((u >> 16) & 1u);       // round-to-nearest-even
    return (unsigned short)(u >> 16);
}
__device__ __forceinline__ float sigm(float x) { return 1.0f / (1.0f + __expf(-x)); }

__device__ __forceinline__ v8f zero8() { v8f z = {0.f,0.f,0.f,0.f,0.f,0.f,0.f,0.f}; return z; }

// A fragment (16x32, MxK) or B fragment (32x16, KxN with weights stored (N,K) row-major):
// lane&15 selects row (M or N), lane>>4 selects K-half; element j -> k = (j>>3)*16 + (lane>>4)*8 + (j&7)
// => two contiguous 16-byte loads per lane.
__device__ __forceinline__ v16bf load_frag(const unsigned short* __restrict__ base,
                                           int row0, int ld, int k0, int lane) {
    const int r  = row0 + (lane & 15);
    const int kh = (lane >> 4) * 8;
    const unsigned short* p = base + (size_t)r * ld + k0 + kh;
    FragU f;
    f.q[0] = *reinterpret_cast<const uint4*>(p);
    f.q[1] = *reinterpret_cast<const uint4*>(p + 16);
    return f.v;
}

#define WMMA_BF16(a, b, c) \
    __builtin_amdgcn_wmma_f32_16x16x32_bf16(false, (a), false, (b), (short)0, (c), false, false)

// ---------------- prep kernels ----------------
__global__ void k_f2bf(const float* __restrict__ src, unsigned short* __restrict__ dst, int n) {
    int i = blockIdx.x * blockDim.x + threadIdx.x;
    if (i < n) dst[i] = f2bf(src[i]);
}

// Wc: [6][H][V] fp32 -> WcT: [6][V][H] bf16 (so B-frag loads are contiguous in K=H)
__global__ void k_transpose_wc(const float* __restrict__ Wc, unsigned short* __restrict__ WcT) {
    int i = blockIdx.x * blockDim.x + threadIdx.x;
    if (i < NCLUS * H_ * V_) {
        int c = i / (H_ * V_);
        int rem = i % (H_ * V_);
        int h = rem / V_;
        int v = rem % V_;
        WcT[(size_t)c * V_ * H_ + (size_t)v * H_ + h] = f2bf(Wc[i]);
    }
}

__global__ void k_bsum(const float* __restrict__ bi, const float* __restrict__ bh,
                       float* __restrict__ bs, int n) {
    int i = blockIdx.x * blockDim.x + threadIdx.x;
    if (i < n) bs[i] = bi[i] + bh[i];
}

__global__ void k_init_state(const float* __restrict__ h0, const float* __restrict__ c0,
                             float* __restrict__ Hst0, float* __restrict__ Hst1,
                             float* __restrict__ Cst0, float* __restrict__ Cst1,
                             unsigned short* __restrict__ Hi0, unsigned short* __restrict__ Hi1) {
    int i = blockIdx.x * blockDim.x + threadIdx.x;
    if (i < B_ * H_) {
        float a = h0[i], b = h0[B_ * H_ + i];
        Hst0[i] = a; Hst1[i] = b;
        Cst0[i] = c0[i]; Cst1[i] = c0[B_ * H_ + i];
        Hi0[i] = f2bf(a); Hi1[i] = f2bf(b);
    }
}

// X[t*B+b] = concat(pc_emb[pc[t,b]], delta_emb[delta[t,b]]) as bf16, ld = 2E = 512
__global__ void k_embed(const int* __restrict__ pc, const int* __restrict__ delta,
                        const float* __restrict__ pc_emb, const float* __restrict__ delta_emb,
                        unsigned short* __restrict__ X) {
    int i = blockIdx.x * blockDim.x + threadIdx.x;
    if (i < TB * (2 * E_)) {
        int row = i >> 9, e = i & 511;
        float v = (e < E_) ? pc_emb[(size_t)pc[row] * E_ + e]
                           : delta_emb[(size_t)delta[row] * E_ + (e - E_)];
        X[i] = f2bf(v);
    }
}

// ---------------- big GEMM: C[M,N] fp32 = A[M,K]bf16 @ W[N,K]bf16^T ----------------
// block = 256 thr = 8 waves; block tile 32x512; wave tile 32x64 (2x4 frags)
// grid = (N/512, M/32)  -- proven spill-free config for the 256-VGPR direct file
__global__ void k_gemm_bf16(const unsigned short* __restrict__ A,
                            const unsigned short* __restrict__ W,
                            float* __restrict__ C, int K, int N) {
    const int lane = threadIdx.x & 31;
    const int wave = threadIdx.x >> 5;
    const int m0 = blockIdx.y * 32;
    const int n0 = blockIdx.x * 512 + wave * 64;
    v8f acc[2][4];
    #pragma unroll
    for (int i = 0; i < 2; i++)
        #pragma unroll
        for (int j = 0; j < 4; j++) acc[i][j] = zero8();

    for (int k0 = 0; k0 < K; k0 += 32) {
        v16bf a[2], b[4];
        #pragma unroll
        for (int i = 0; i < 2; i++) a[i] = load_frag(A, m0 + 16 * i, K, k0, lane);
        #pragma unroll
        for (int j = 0; j < 4; j++) b[j] = load_frag(W, n0 + 16 * j, K, k0, lane);
        #pragma unroll
        for (int i = 0; i < 2; i++)
            #pragma unroll
            for (int j = 0; j < 4; j++) acc[i][j] = WMMA_BF16(a[i], b[j], acc[i][j]);
    }
    const int half = lane >> 4, co = lane & 15;
    #pragma unroll
    for (int i = 0; i < 2; i++) {
        const int rb = m0 + 16 * i + 8 * half;
        #pragma unroll
        for (int j = 0; j < 4; j++) {
            const int col = n0 + 16 * j + co;
            #pragma unroll
            for (int r = 0; r < 8; r++) C[(size_t)(rb + r) * N + col] = acc[i][j][r];
        }
    }
}

// ---------------- fused recurrent step: gates GEMM + LSTM cell ----------------
// grid = 32 blocks (16 H-cols each), 512 thr = 16 waves.
// wave -> (kh = wave>>3, g = (wave>>1)&3, mt = wave&1): K split in two halves of 256
// to halve the dependent-WMMA critical path; halves summed in LDS during the cell update.
__global__ void k_lstm_step(const unsigned short* __restrict__ Hprev,  // [32][512] bf16
                            const unsigned short* __restrict__ Whh,    // [2048][512] bf16
                            const float* __restrict__ Gin,             // [32][2048] this t
                            const float* __restrict__ bsum,            // [2048]
                            float* __restrict__ Cst,                   // [32][512]
                            float* __restrict__ Hst,                   // [32][512]
                            unsigned short* __restrict__ Hout) {       // [32][512] bf16 = Y[t]
    __shared__ float gl[2][4][B_][16];
    const int lane = threadIdx.x & 31;
    const int wave = threadIdx.x >> 5;
    const int kh = wave >> 3;
    const int g = (wave >> 1) & 3, mt = wave & 1;
    const int j0 = blockIdx.x * 16;
    const int n0 = g * H_ + j0;
    const int kbeg = kh * (H_ / 2), kend = kbeg + (H_ / 2);

    v8f acc = zero8();
    for (int k0 = kbeg; k0 < kend; k0 += 32) {
        if (k0 + 32 < kend) {  // prefetch next K-tile of the streamed weight matrix
            __builtin_prefetch(Whh + (size_t)(n0 + (lane & 15)) * H_ + k0 + 32, 0, 0);
        }
        v16bf a = load_frag(Hprev, mt * 16, H_, k0, lane);
        v16bf b = load_frag(Whh, n0, H_, k0, lane);
        acc = WMMA_BF16(a, b, acc);
    }
    const int half = lane >> 4, co = lane & 15;
    #pragma unroll
    for (int r = 0; r < 8; r++) {
        const int brow = mt * 16 + 8 * half + r;
        gl[kh][g][brow][co] = acc[r];
    }
    __syncthreads();
    {
        const int e = threadIdx.x;          // 512 threads, one (b, jl) element each
        const int b = e >> 4, jl = e & 15;
        const int jn = j0 + jl;
        const float iv = gl[0][0][b][jl] + gl[1][0][b][jl] + Gin[b * G4H + 0 * H_ + jn] + bsum[0 * H_ + jn];
        const float fv = gl[0][1][b][jl] + gl[1][1][b][jl] + Gin[b * G4H + 1 * H_ + jn] + bsum[1 * H_ + jn];
        const float gv = gl[0][2][b][jl] + gl[1][2][b][jl] + Gin[b * G4H + 2 * H_ + jn] + bsum[2 * H_ + jn];
        const float ov = gl[0][3][b][jl] + gl[1][3][b][jl] + Gin[b * G4H + 3 * H_ + jn] + bsum[3 * H_ + jn];
        const int idx = b * H_ + jn;
        float c = sigm(fv) * Cst[idx] + sigm(iv) * tanhf(gv);
        float h = sigm(ov) * tanhf(c);
        Cst[idx] = c;
        Hst[idx] = h;
        Hout[idx] = f2bf(h);
    }
}

// ---------------- cluster-routed logits GEMM ----------------
// grid = (V/512, T); block tile 32x512 of logits[t]; wave tile 32x64
__global__ void k_logits(const unsigned short* __restrict__ Y1,   // [TB][512] bf16
                         const unsigned short* __restrict__ WcT,  // [6][V][H] bf16
                         const float* __restrict__ bc,            // [6][V]
                         const int* __restrict__ clusters,        // [T]
                         float* __restrict__ Lg) {                // [TB][V]
    const int t = blockIdx.y;
    const int cl = clusters[t];
    const unsigned short* A = Y1 + (size_t)t * B_ * H_;
    const unsigned short* W = WcT + (size_t)cl * V_ * H_;
    const int lane = threadIdx.x & 31;
    const int wave = threadIdx.x >> 5;
    const int n0 = blockIdx.x * 512 + wave * 64;

    v8f acc[2][4];
    #pragma unroll
    for (int i = 0; i < 2; i++)
        #pragma unroll
        for (int j = 0; j < 4; j++) acc[i][j] = zero8();

    for (int k0 = 0; k0 < H_; k0 += 32) {
        v16bf a[2], b[4];
        #pragma unroll
        for (int i = 0; i < 2; i++) a[i] = load_frag(A, 16 * i, H_, k0, lane);
        #pragma unroll
        for (int j = 0; j < 4; j++) b[j] = load_frag(W, n0 + 16 * j, H_, k0, lane);
        #pragma unroll
        for (int i = 0; i < 2; i++)
            #pragma unroll
            for (int j = 0; j < 4; j++) acc[i][j] = WMMA_BF16(a[i], b[j], acc[i][j]);
    }
    const int half = lane >> 4, co = lane & 15;
    #pragma unroll
    for (int i = 0; i < 2; i++) {
        const int rb = 16 * i + 8 * half;
        #pragma unroll
        for (int j = 0; j < 4; j++) {
            const int col = n0 + 16 * j + co;
            const float bv = bc[cl * V_ + col];
            #pragma unroll
            for (int r = 0; r < 8; r++)
                Lg[((size_t)t * B_ + rb + r) * V_ + col] = acc[i][j][r] + bv;
        }
    }
}

// ---------------- log-softmax + top-10 + per-row NLL ----------------
__global__ void k_softmax_topk(const float* __restrict__ Lg, const int* __restrict__ target,
                               float* __restrict__ preds_out, float* __restrict__ rowloss) {
    const int row = blockIdx.x;
    const float* x = Lg + (size_t)row * V_;
    __shared__ float sv[256];
    __shared__ int   si[256];
    __shared__ int   excl[NPRED];
    __shared__ float smax, ssum;
    const int tid = threadIdx.x;

    float m = -3.4e38f;
    for (int i = tid; i < V_; i += 256) m = fmaxf(m, x[i]);
    sv[tid] = m; __syncthreads();
    for (int s = 128; s > 0; s >>= 1) { if (tid < s) sv[tid] = fmaxf(sv[tid], sv[tid + s]); __syncthreads(); }
    if (tid == 0) smax = sv[0];
    __syncthreads();
    const float mm = smax;

    float su = 0.f;
    for (int i = tid; i < V_; i += 256) su += __expf(x[i] - mm);
    sv[tid] = su; __syncthreads();
    for (int s = 128; s > 0; s >>= 1) { if (tid < s) sv[tid] += sv[tid + s]; __syncthreads(); }
    if (tid == 0) ssum = sv[0];
    __syncthreads();
    const float lse = mm + __logf(ssum);
    if (tid == 0) rowloss[row] = -(x[target[row]] - lse);

    for (int p = 0; p < NPRED; p++) {
        float bm = -3.4e38f; int bi = V_;
        for (int i = tid; i < V_; i += 256) {
            bool ex = false;
            for (int q = 0; q < p; q++) ex |= (excl[q] == i);
            const float v = x[i];
            if (!ex && (v > bm || (v == bm && i < bi))) { bm = v; bi = i; }
        }
        sv[tid] = bm; si[tid] = bi; __syncthreads();
        for (int s = 128; s > 0; s >>= 1) {
            if (tid < s) {
                if (sv[tid + s] > sv[tid] || (sv[tid + s] == sv[tid] && si[tid + s] < si[tid])) {
                    sv[tid] = sv[tid + s]; si[tid] = si[tid + s];
                }
            }
            __syncthreads();
        }
        if (tid == 0) { excl[p] = si[0]; preds_out[(size_t)row * NPRED + p] = (float)si[0]; }
        __syncthreads();
    }
}

// deterministic fixed-order loss reduction (no float atomics)
__global__ void k_loss(const float* __restrict__ rowloss, float* __restrict__ out0) {
    __shared__ float sv[256];
    const int tid = threadIdx.x;
    float s = 0.f;
    for (int i = tid; i < TB; i += 256) s += rowloss[i];
    sv[tid] = s; __syncthreads();
    for (int st = 128; st > 0; st >>= 1) { if (tid < st) sv[tid] += sv[tid + st]; __syncthreads(); }
    if (tid == 0) out0[0] = sv[0] / (float)TB;
}

__global__ void k_final(const float* __restrict__ Hst0, const float* __restrict__ Hst1,
                        const float* __restrict__ Cst0, const float* __restrict__ Cst1,
                        float* __restrict__ out) {
    int i = blockIdx.x * blockDim.x + threadIdx.x;
    if (i < B_ * H_) {
        const int base = 1 + TB * NPRED;
        out[base + i] = Hst0[i];
        out[base + B_ * H_ + i] = Hst1[i];
        out[base + 2 * B_ * H_ + i] = Cst0[i];
        out[base + 3 * B_ * H_ + i] = Cst1[i];
    }
}

// ---------------- host launcher ----------------
extern "C" void kernel_launch(void* const* d_in, const int* in_sizes, int n_in,
                              void* d_out, int out_size, void* d_ws, size_t ws_size,
                              hipStream_t stream) {
    (void)in_sizes; (void)n_in; (void)out_size; (void)ws_size;
    const int*   pc        = (const int*)d_in[0];
    const int*   delta     = (const int*)d_in[1];
    const int*   clusters  = (const int*)d_in[2];
    const int*   target    = (const int*)d_in[3];
    const float* h0        = (const float*)d_in[4];
    const float* c0        = (const float*)d_in[5];
    const float* pc_emb    = (const float*)d_in[6];
    const float* delta_emb = (const float*)d_in[7];
    const float* w_ih0     = (const float*)d_in[8];
    const float* w_hh0     = (const float*)d_in[9];
    const float* b_ih0     = (const float*)d_in[10];
    const float* b_hh0     = (const float*)d_in[11];
    const float* w_ih1     = (const float*)d_in[12];
    const float* w_hh1     = (const float*)d_in[13];
    const float* b_ih1     = (const float*)d_in[14];
    const float* b_hh1     = (const float*)d_in[15];
    const float* Wc        = (const float*)d_in[16];
    const float* bc        = (const float*)d_in[17];
    float* out = (float*)d_out;

    char* wsb = (char*)d_ws;
    size_t off = 0;
    auto alloc = [&](size_t bytes) -> char* {
        char* p = wsb + off;
        off += (bytes + 255) & ~(size_t)255;
        return p;
    };
    unsigned short* Xb    = (unsigned short*)alloc((size_t)TB * 512 * 2);
    unsigned short* Y0    = (unsigned short*)alloc((size_t)TB * 512 * 2);
    unsigned short* Y1    = (unsigned short*)alloc((size_t)TB * 512 * 2);
    float*          Gin   = (float*)alloc((size_t)TB * G4H * 4);   // reused: Gin0 -> Gin1 -> logits
    unsigned short* wih0b = (unsigned short*)alloc((size_t)G4H * 512 * 2);
    unsigned short* whh0b = (unsigned short*)alloc((size_t)G4H * 512 * 2);
    unsigned short* wih1b = (unsigned short*)alloc((size_t)G4H * 512 * 2);
    unsigned short* whh1b = (unsigned short*)alloc((size_t)G4H * 512 * 2);
    unsigned short* WcTb  = (unsigned short*)alloc((size_t)NCLUS * V_ * H_ * 2);
    float*          bsum0 = (float*)alloc(G4H * 4);
    float*          bsum1 = (float*)alloc(G4H * 4);
    unsigned short* Hi0   = (unsigned short*)alloc(B_ * H_ * 2);
    unsigned short* Hi1   = (unsigned short*)alloc(B_ * H_ * 2);
    float*          Hst0  = (float*)alloc(B_ * H_ * 4);
    float*          Hst1  = (float*)alloc(B_ * H_ * 4);
    float*          Cst0  = (float*)alloc(B_ * H_ * 4);
    float*          Cst1  = (float*)alloc(B_ * H_ * 4);
    float*          rloss = (float*)alloc(TB * 4);

    const int TPB = 256;
    const int WN = G4H * 512;  // per-weight elements
    k_f2bf<<<(WN + TPB - 1) / TPB, TPB, 0, stream>>>(w_ih0, wih0b, WN);
    k_f2bf<<<(WN + TPB - 1) / TPB, TPB, 0, stream>>>(w_hh0, whh0b, WN);
    k_f2bf<<<(WN + TPB - 1) / TPB, TPB, 0, stream>>>(w_ih1, wih1b, WN);
    k_f2bf<<<(WN + TPB - 1) / TPB, TPB, 0, stream>>>(w_hh1, whh1b, WN);
    k_transpose_wc<<<(NCLUS * H_ * V_ + TPB - 1) / TPB, TPB, 0, stream>>>(Wc, WcTb);
    k_bsum<<<(G4H + TPB - 1) / TPB, TPB, 0, stream>>>(b_ih0, b_hh0, bsum0, G4H);
    k_bsum<<<(G4H + TPB - 1) / TPB, TPB, 0, stream>>>(b_ih1, b_hh1, bsum1, G4H);
    k_init_state<<<(B_ * H_ + TPB - 1) / TPB, TPB, 0, stream>>>(h0, c0, Hst0, Hst1, Cst0, Cst1, Hi0, Hi1);
    k_embed<<<(TB * 512 + TPB - 1) / TPB, TPB, 0, stream>>>(pc, delta, pc_emb, delta_emb, Xb);

    // layer 0: hoisted input projection, then fused recurrent steps
    k_gemm_bf16<<<dim3(G4H / 512, TB / 32), 256, 0, stream>>>(Xb, wih0b, Gin, 512, G4H);
    for (int t = 0; t < T_; t++) {
        const unsigned short* Hp = (t == 0) ? Hi0 : (Y0 + (size_t)(t - 1) * B_ * H_);
        k_lstm_step<<<H_ / 16, 512, 0, stream>>>(Hp, whh0b, Gin + (size_t)t * B_ * G4H,
                                                 bsum0, Cst0, Hst0, Y0 + (size_t)t * B_ * H_);
    }
    // layer 1
    k_gemm_bf16<<<dim3(G4H / 512, TB / 32), 256, 0, stream>>>(Y0, wih1b, Gin, 512, G4H);
    for (int t = 0; t < T_; t++) {
        const unsigned short* Hp = (t == 0) ? Hi1 : (Y1 + (size_t)(t - 1) * B_ * H_);
        k_lstm_step<<<H_ / 16, 512, 0, stream>>>(Hp, whh1b, Gin + (size_t)t * B_ * G4H,
                                                 bsum1, Cst1, Hst1, Y1 + (size_t)t * B_ * H_);
    }
    // cluster-routed logits (reuse Gin buffer as logits [TB][V])
    k_logits<<<dim3(V_ / 512, T_), 256, 0, stream>>>(Y1, WcTb, bc, clusters, Gin);
    // log-softmax + top-10 + per-row loss
    k_softmax_topk<<<TB, 256, 0, stream>>>(Gin, target, out + 1, rloss);
    k_loss<<<1, 256, 0, stream>>>(rloss, out);
    k_final<<<(B_ * H_ + TPB - 1) / TPB, TPB, 0, stream>>>(Hst0, Hst1, Cst0, Cst1, out);
}
